// MOE_10187662426435
// MI455X (gfx1250) — compile-verified, matrix operand order checked by reading, not run on hardware
//
#include <hip/hip_runtime.h>
#include <hip/hip_bf16.h>
#include <stdint.h>

// MoE top-2 routed FFN for MI455X (gfx1250, wave32, WMMA bf16, async-to-LDS).
// Workspace (~291 MB):
//   counts[8] | top1[8] | probsum[8] | base[8] | slot_tok[8*4096] | slot_w[8*4096]
//   xg   bf16 [8704][1024]    gathered tokens (per-expert rows padded to 64)
//   hbuf bf16 [8704][4096]    gelu(gate)*up staging
//   w1b  bf16 [8][8192][1024] pre-converted weights
//   w2b  bf16 [8][1024][4096]

#define T_TOK 4096
#define H_DIM 1024
#define E_EXP 8
#define F_DIM 4096
#define ROWS_MAX 8704

typedef __attribute__((ext_vector_type(16))) __bf16   v16bf;
typedef __attribute__((ext_vector_type(8)))  float    v8f;
typedef __attribute__((ext_vector_type(8)))  unsigned v8u;

__device__ __forceinline__ unsigned short f2bf(float f) {
    unsigned u = __float_as_uint(f);
    u = (u + 0x7fffu + ((u >> 16) & 1u)) >> 16;   // RNE
    return (unsigned short)u;
}
__device__ __forceinline__ unsigned pack2(float lo, float hi) {
    return (unsigned)f2bf(lo) | ((unsigned)f2bf(hi) << 16);
}
// gelu_tanh(x) = x * sigmoid(2*0.7978845608*(x + 0.044715 x^3)) -- branchless
__device__ __forceinline__ float gelu_t(float x) {
    float u = 1.5957691216057308f * (x + 0.044715f * x * x * x);
    return x * (1.f / (1.f + __expf(-u)));
}

// ---- CDNA5 async global->LDS (ASYNCcnt) --------------------------------
__device__ __forceinline__ void async_ld16(unsigned lds_off, const void* gaddr) {
    asm volatile("global_load_async_to_lds_b128 %0, %1, off"
                 :: "v"(lds_off), "v"((unsigned long long)(uintptr_t)gaddr)
                 : "memory");
}
__device__ __forceinline__ void wait_async0() { asm volatile("s_wait_asynccnt 0x0" ::: "memory"); }
__device__ __forceinline__ void wait_async4() { asm volatile("s_wait_asynccnt 0x4" ::: "memory"); }
__device__ __forceinline__ void wait_async6() { asm volatile("s_wait_asynccnt 0x6" ::: "memory"); }

// 16x32 bf16 fragment from LDS tile stored [row][ld] row-major.
// Lane<16 -> K 0..7 & 16..23 ; lane>=16 -> K 8..15 & 24..31 (ISA 16-bit layout).
__device__ __forceinline__ v16bf ld_frag(const unsigned short* tile, int ld) {
    const int lane = threadIdx.x & 31;
    const int row  = lane & 15;
    const int hi   = lane >> 4;
    const unsigned* p = (const unsigned*)(tile + row * ld + hi * 8);
    union { v8u u; v16bf h; } c;
    c.u[0] = p[0]; c.u[1] = p[1]; c.u[2] = p[2];  c.u[3] = p[3];
    c.u[4] = p[8]; c.u[5] = p[9]; c.u[6] = p[10]; c.u[7] = p[11]; // +16 elems
    return c.h;
}

// ---- small kernels ------------------------------------------------------
__global__ void moe_zero_meta(int* counts, int* top1, float* probsum) {
    int i = threadIdx.x;
    if (i < E_EXP) { counts[i] = 0; top1[i] = 0; probsum[i] = 0.f; }
}

__global__ void moe_zero_out(float* out) {
    size_t i = (size_t)blockIdx.x * blockDim.x + threadIdx.x;
    float4 z = {0.f, 0.f, 0.f, 0.f};
    ((float4*)out)[i] = z;
}

// fp32 -> bf16 bulk convert, 8 elements/thread
__global__ void moe_cvt(const float* __restrict__ src, unsigned short* __restrict__ dst) {
    size_t i = (size_t)blockIdx.x * blockDim.x + threadIdx.x;
    const float4* s = (const float4*)src + i * 2;
    float4 a = s[0], b = s[1];
    uint4 pk; pk.x = pack2(a.x, a.y); pk.y = pack2(a.z, a.w);
              pk.z = pack2(b.x, b.y); pk.w = pack2(b.z, b.w);
    ((uint4*)dst)[i] = pk;
}

// One wave per token: logits, softmax, top-2, routing lists, loss stats.
__global__ void moe_gate(const float* __restrict__ x, const float* __restrict__ gw,
                         const float* __restrict__ gb,
                         int* counts, int* top1, float* probsum,
                         int* slot_tok, float* slot_w) {
    const int lane = threadIdx.x & 31;
    const int t = blockIdx.x * 8 + (threadIdx.x >> 5);
    const float* xr = x + (size_t)t * H_DIM;
    float acc[E_EXP];
#pragma unroll
    for (int e = 0; e < E_EXP; ++e) acc[e] = 0.f;
    for (int k = lane; k < H_DIM; k += 32) {
        float xv = xr[k];
#pragma unroll
        for (int e = 0; e < E_EXP; ++e) acc[e] += xv * gw[e * H_DIM + k];
    }
#pragma unroll
    for (int e = 0; e < E_EXP; ++e) {
#pragma unroll
        for (int off = 16; off >= 1; off >>= 1) acc[e] += __shfl_xor(acc[e], off, 32);
    }
    if (lane == 0) {
        float lg[E_EXP], mx = -1e30f;
        for (int e = 0; e < E_EXP; ++e) { lg[e] = acc[e] + gb[e]; mx = fmaxf(mx, lg[e]); }
        float s = 0.f, p[E_EXP];
        for (int e = 0; e < E_EXP; ++e) { p[e] = __expf(lg[e] - mx); s += p[e]; }
        float inv = 1.f / s;
        int i0 = 0;
        for (int e = 1; e < E_EXP; ++e) if (p[e] > p[i0]) i0 = e;
        int i1 = (i0 == 0) ? 1 : 0;
        for (int e = 0; e < E_EXP; ++e) if (e != i0 && p[e] > p[i1]) i1 = e;
        for (int e = 0; e < E_EXP; ++e) atomicAdd(&probsum[e], p[e] * inv);
        atomicAdd(&top1[i0], 1);
        float v0 = p[i0] * inv, v1 = p[i1] * inv;
        float tot = v0 + v1, w0 = v0 / tot, w1 = v1 / tot;
        int s0 = atomicAdd(&counts[i0], 1);
        slot_tok[i0 * T_TOK + s0] = t; slot_w[i0 * T_TOK + s0] = w0;
        int s1 = atomicAdd(&counts[i1], 1);
        slot_tok[i1 * T_TOK + s1] = t; slot_w[i1 * T_TOK + s1] = w1;
    }
}

__global__ void moe_finalize(const int* counts, const int* top1, const float* probsum,
                             int* base, float* loss_out) {
    if (threadIdx.x == 0) {
        int b = 0; float s = 0.f;
        for (int e = 0; e < E_EXP; ++e) {
            base[e] = b;
            b += ((counts[e] + 63) >> 6) << 6;
            s += probsum[e] * (float)top1[e];
        }
        *loss_out = s * 8.f / ((float)T_TOK * (float)T_TOK);  // mean_e(p*f)*E*E
    }
}

__global__ void moe_gather(const float* __restrict__ x, const int* counts, const int* base,
                           const int* slot_tok, unsigned short* __restrict__ xg) {
    const int e = blockIdx.y, s = blockIdx.x;
    if (s >= counts[e]) return;
    const int tok = slot_tok[e * T_TOK + s];
    const float* src = x + (size_t)tok * H_DIM;
    unsigned short* dst = xg + (size_t)(base[e] + s) * H_DIM;
    for (int k = threadIdx.x * 4; k < H_DIM; k += blockDim.x * 4) {
        float4 v = *(const float4*)(src + k);
        uint2 pk; pk.x = pack2(v.x, v.y); pk.y = pack2(v.z, v.w);
        *(uint2*)(dst + k) = pk;
    }
}

// Issue one 64x64 bf16 tile as 512 b128 async transfers (2 per thread).
__device__ __forceinline__ void tile_async(const unsigned short* gsrc, size_t grow_stride,
                                           unsigned lds_base, int tid) {
#pragma unroll
    for (int i = 0; i < 2; ++i) {
        int idx = tid * 2 + i;
        int row = idx >> 3, seg = (idx & 7) * 8;
        async_ld16(lds_base + (unsigned)((row * 64 + seg) * 2),
                   gsrc + (size_t)row * grow_stride + seg);
    }
}

// ---- Phase 1: GU = Xe @ W1e^T (+b1); h = gelu(gate)*up -> hbuf ---------
// 64x64 block tile, KC=64 double-buffered, all operands async bf16.
__global__ void __launch_bounds__(256)
moe_ffn1(const unsigned short* __restrict__ xg,
         const unsigned short* __restrict__ w1b, const float* __restrict__ b1,
         const int* counts, const int* base,
         unsigned short* __restrict__ hbuf) {
    const int e  = blockIdx.z;
    const int mt = blockIdx.y;
    const int nt = blockIdx.x;
    const int cnt = counts[e];
    if (mt * 64 >= cnt) return;
    const int rowbase = base[e] + mt * 64;
    const int n0 = nt * 64;
    const unsigned short* w1e = w1b + (size_t)e * ((size_t)2 * F_DIM * H_DIM);
    const float* b1e = b1 + (size_t)e * (2 * F_DIM);

    extern __shared__ unsigned short smem[];   // 2 buffers x (A|Bg|Bu) 64x64 bf16
    const int tid = threadIdx.x;
    const int wv = tid >> 5, lane = tid & 31, hi = lane >> 4;
    const int pr0 = wv >> 2, pc = wv & 3;

    v8f cg[2] = {};
    v8f cu[2] = {};

    const unsigned short* ag = xg  + (size_t)rowbase * H_DIM;
    const unsigned short* bg = w1e + (size_t)n0 * H_DIM;
    const unsigned short* bu = w1e + (size_t)(F_DIM + n0) * H_DIM;

    // prologue: fill buffer 0
    tile_async(ag, H_DIM, 0u,       tid);
    tile_async(bg, H_DIM, 8192u,    tid);
    tile_async(bu, H_DIM, 16384u,   tid);

    const int NIT = H_DIM / 64;
    for (int it = 0; it < NIT; ++it) {
        const int cur = it & 1;
        if (it + 1 < NIT) {   // prefetch next buffer, then wait for current 6
            const unsigned nb = (it + 1) & 1 ? 24576u : 0u;
            const int k1 = (it + 1) * 64;
            tile_async(ag + k1, H_DIM, nb,          tid);
            tile_async(bg + k1, H_DIM, nb + 8192u,  tid);
            tile_async(bu + k1, H_DIM, nb + 16384u, tid);
            wait_async6();
        } else {
            wait_async0();
        }
        __syncthreads();
        const unsigned short* At = smem + cur * 12288;
        const unsigned short* Bg = At + 4096;
        const unsigned short* Bu = At + 8192;
#pragma unroll
        for (int c = 0; c < 2; ++c) {            // two K=32 chunks per buffer
            v16bf fbg = ld_frag(Bg + pc * 16 * 64 + c * 32, 64);
            v16bf fbu = ld_frag(Bu + pc * 16 * 64 + c * 32, 64);
#pragma unroll
            for (int t = 0; t < 2; ++t) {
                v16bf fa = ld_frag(At + (pr0 + 2 * t) * 16 * 64 + c * 32, 64);
                cg[t] = __builtin_amdgcn_wmma_f32_16x16x32_bf16(false, fa, false, fbg, (short)0, cg[t], false, false);
                cu[t] = __builtin_amdgcn_wmma_f32_16x16x32_bf16(false, fa, false, fbu, (short)0, cu[t], false, false);
            }
        }
        __syncthreads();
    }
    const int n = n0 + pc * 16 + (lane & 15);
    const float bgb = b1e[n];
    const float bub = b1e[F_DIM + n];
#pragma unroll
    for (int t = 0; t < 2; ++t) {
        const int pr = pr0 + 2 * t;
#pragma unroll
        for (int j = 0; j < 8; ++j) {
            int m = pr * 16 + j + 8 * hi;
            float hval = gelu_t(cg[t][j] + bgb) * (cu[t][j] + bub);
            hbuf[(size_t)(rowbase + m) * F_DIM + n] = f2bf(hval);
        }
    }
}

// ---- Phase 2: Y = h @ W2e^T (+b2); out[token] += weight * Y ------------
__global__ void __launch_bounds__(256)
moe_ffn2(const unsigned short* __restrict__ hbuf,
         const unsigned short* __restrict__ w2b, const float* __restrict__ b2,
         const int* counts, const int* base,
         const int* slot_tok, const float* slot_w,
         float* __restrict__ out) {
    const int e  = blockIdx.z;
    const int mt = blockIdx.y;
    const int nt = blockIdx.x;
    const int cnt = counts[e];
    if (mt * 64 >= cnt) return;
    const int rowbase = base[e] + mt * 64;
    const int n0 = nt * 64;
    const unsigned short* w2e = w2b + (size_t)e * ((size_t)H_DIM * F_DIM);
    const float* b2e = b2 + (size_t)e * H_DIM;

    extern __shared__ unsigned short smem[];   // 2 buffers x (A|B) 64x64 bf16
    const int tid = threadIdx.x;
    const int wv = tid >> 5, lane = tid & 31, hi = lane >> 4;
    const int pr0 = wv >> 2, pc = wv & 3;

    v8f y[2] = {};

    const unsigned short* ag = hbuf + (size_t)rowbase * F_DIM;
    const unsigned short* bg = w2e  + (size_t)n0 * F_DIM;

    tile_async(ag, F_DIM, 0u,    tid);
    tile_async(bg, F_DIM, 8192u, tid);

    const int NIT = F_DIM / 64;
    for (int it = 0; it < NIT; ++it) {
        const int cur = it & 1;
        if (it + 1 < NIT) {
            const unsigned nb = (it + 1) & 1 ? 16384u : 0u;
            const int k1 = (it + 1) * 64;
            tile_async(ag + k1, F_DIM, nb,         tid);
            tile_async(bg + k1, F_DIM, nb + 8192u, tid);
            wait_async4();
        } else {
            wait_async0();
        }
        __syncthreads();
        const unsigned short* At = smem + cur * 8192;
        const unsigned short* Bt = At + 4096;
#pragma unroll
        for (int c = 0; c < 2; ++c) {
            v16bf fb = ld_frag(Bt + pc * 16 * 64 + c * 32, 64);
#pragma unroll
            for (int t = 0; t < 2; ++t) {
                v16bf fa = ld_frag(At + (pr0 + 2 * t) * 16 * 64 + c * 32, 64);
                y[t] = __builtin_amdgcn_wmma_f32_16x16x32_bf16(false, fa, false, fb, (short)0, y[t], false, false);
            }
        }
        __syncthreads();
    }
    const int n = n0 + pc * 16 + (lane & 15);
    const float bb = b2e[n];
#pragma unroll
    for (int t = 0; t < 2; ++t) {
        const int pr = pr0 + 2 * t;
#pragma unroll
        for (int j = 0; j < 8; ++j) {
            int s = mt * 64 + pr * 16 + j + 8 * hi;
            if (s < cnt) {
                int tok = slot_tok[e * T_TOK + s];
                float wgt = slot_w[e * T_TOK + s];
                atomicAdd(out + (size_t)tok * H_DIM + n, (y[t][j] + bb) * wgt);
            }
        }
    }
}

extern "C" void kernel_launch(void* const* d_in, const int* in_sizes, int n_in,
                              void* d_out, int out_size, void* d_ws, size_t ws_size,
                              hipStream_t stream) {
    (void)in_sizes; (void)n_in; (void)out_size; (void)ws_size;
    const float* x  = (const float*)d_in[0];
    const float* gw = (const float*)d_in[1];
    const float* gb = (const float*)d_in[2];
    const float* w1 = (const float*)d_in[3];
    const float* b1 = (const float*)d_in[4];
    const float* w2 = (const float*)d_in[5];
    const float* b2 = (const float*)d_in[6];
    float* out = (float*)d_out;

    char* ws = (char*)d_ws;
    int*   counts   = (int*)ws;
    int*   top1     = counts + 8;
    float* probsum  = (float*)(top1 + 8);
    int*   base     = (int*)(probsum + 8);
    int*   slot_tok = base + 8;
    float* slot_w   = (float*)(slot_tok + E_EXP * T_TOK);
    uintptr_t p = ((uintptr_t)(slot_w + E_EXP * T_TOK) + 255) & ~(uintptr_t)255;
    unsigned short* xg   = (unsigned short*)p;
    unsigned short* hbuf = xg   + (size_t)ROWS_MAX * H_DIM;
    unsigned short* w1b  = hbuf + (size_t)ROWS_MAX * F_DIM;
    unsigned short* w2b  = w1b  + (size_t)E_EXP * 2 * F_DIM * H_DIM;

    const size_t n_w1 = (size_t)E_EXP * 2 * F_DIM * H_DIM;   // 67,108,864
    const size_t n_w2 = (size_t)E_EXP * H_DIM * F_DIM;       // 33,554,432

    hipLaunchKernelGGL(moe_zero_meta, dim3(1), dim3(32), 0, stream, counts, top1, probsum);
    hipLaunchKernelGGL(moe_zero_out, dim3(4096), dim3(256), 0, stream, out);
    hipLaunchKernelGGL(moe_cvt, dim3((unsigned)(n_w1 / 2048)), dim3(256), 0, stream, w1, w1b);
    hipLaunchKernelGGL(moe_cvt, dim3((unsigned)(n_w2 / 2048)), dim3(256), 0, stream, w2, w2b);
    hipLaunchKernelGGL(moe_gate, dim3(T_TOK / 8), dim3(256), 0, stream,
                       x, gw, gb, counts, top1, probsum, slot_tok, slot_w);
    hipLaunchKernelGGL(moe_finalize, dim3(1), dim3(32), 0, stream,
                       counts, top1, probsum, base, out + (size_t)T_TOK * H_DIM);
    hipLaunchKernelGGL(moe_gather, dim3(T_TOK, E_EXP), dim3(256), 0, stream,
                       x, counts, base, slot_tok, xg);
    hipLaunchKernelGGL(moe_ffn1, dim3(64, 64, 8), dim3(256), 2 * 3 * 64 * 64 * 2, stream,
                       xg, w1b, b1, counts, base, hbuf);
    hipLaunchKernelGGL(moe_ffn2, dim3(16, 64, 8), dim3(256), 2 * 2 * 64 * 64 * 2, stream,
                       hbuf, w2b, b2, counts, base, slot_tok, slot_w, out);
}
